// KANSplineLayer_12876311954298
// MI455X (gfx1250) — matrix operand order repaired and checked by reading, not compiled, not run on hardware
//
#include <hip/hip_runtime.h>
#include <hip/hip_bf16.h>
#include <math.h>

// ---- problem constants (from reference) ----
#define BATCH   16384
#define IN_DIM  512
#define OUT_DIM 1024
#define NCOEF   8          // grid cols (12) - K(3) - 1
#define NKNOT   12
#define KAUG    (IN_DIM * NCOEF + IN_DIM)   // 4096 spline + 512 base = 4608
#define KT_STEPS (KAUG / 32)                // 144

typedef __attribute__((ext_vector_type(16))) _Float16 v16h;
typedef __attribute__((ext_vector_type(8)))  _Float16 v8h;
typedef __attribute__((ext_vector_type(8)))  float    v8f;

// =====================================================================
// Kernel 1: activations.  One thread per (b, i).
// Writes A[b][i*8 + k] = B-spline_k(x), A[b][4096 + i] = silu(x), as f16.
// =====================================================================
__global__ __launch_bounds__(256) void kan_act_kernel(
    const float* __restrict__ x,
    const float* __restrict__ grid,
    _Float16* __restrict__ A)
{
    int gid = blockIdx.x * 256 + threadIdx.x;
    if (gid >= BATCH * IN_DIM) return;
    int b = gid >> 9;          // / IN_DIM
    int i = gid & (IN_DIM - 1);

    float xv = x[(size_t)b * IN_DIM + i];

    float t[NKNOT];
#pragma unroll
    for (int j = 0; j < NKNOT; ++j) t[j] = grid[i * NKNOT + j];

    // Cox–de Boor, degree 0..3.  Uniform grid => denominators nonzero.
    float v[NKNOT - 1];
#pragma unroll
    for (int j = 0; j < NKNOT - 1; ++j)
        v[j] = (xv >= t[j] && xv < t[j + 1]) ? 1.0f : 0.0f;
#pragma unroll
    for (int ord = 1; ord <= 3; ++ord) {
#pragma unroll
        for (int j = 0; j < NKNOT - 2; ++j) {
            if (j < NKNOT - 1 - ord) {
                float left  = (xv - t[j]) / (t[j + ord] - t[j]);
                float right = (t[j + ord + 1] - xv) / (t[j + ord + 1] - t[j + 1]);
                v[j] = left * v[j] + right * v[j + 1];
            }
        }
    }

    // pack 8 f16 basis values -> one 16B store (row base is 16B aligned)
    union { _Float16 h[8]; uint4 u; } pk;
#pragma unroll
    for (int k = 0; k < NCOEF; ++k) pk.h[k] = (_Float16)v[k];
    *(uint4*)(A + (size_t)b * KAUG + (size_t)i * NCOEF) = pk.u;

    float s = xv / (1.0f + __expf(-xv));          // silu
    A[(size_t)b * KAUG + IN_DIM * NCOEF + i] = (_Float16)s;
}

// =====================================================================
// Kernel 2: pack weights.  Wt[n][r] (N-major, f16), r = i*8+k for splines,
// r = 4096+i for the base term.
// =====================================================================
__global__ __launch_bounds__(256) void kan_pack_kernel(
    const float* __restrict__ coef,        // [IN][OUT][8]
    const float* __restrict__ scale_base,  // [IN][OUT]
    _Float16* __restrict__ Wt)             // [OUT][KAUG]
{
    int gid = blockIdx.x * 256 + threadIdx.x;
    if (gid >= OUT_DIM * KAUG) return;
    int n = gid / KAUG;
    int r = gid - n * KAUG;
    float val;
    if (r < IN_DIM * NCOEF) {
        int i = r >> 3;
        int k = r & 7;
        val = coef[((size_t)i * OUT_DIM + n) * NCOEF + k];
    } else {
        int i = r - IN_DIM * NCOEF;
        val = scale_base[(size_t)i * OUT_DIM + n];
    }
    Wt[(size_t)n * KAUG + r] = (_Float16)val;
}

// =====================================================================
// Kernel 3: WMMA GEMM with async global->LDS double buffering.
// out[16384][1024] = A[16384][4608] * Wt^T
// Block: 256 threads (8 waves), tile 128(M) x 128(N), k-step 32.
// Wave grid 2x4; each wave: 4 M-tiles x 2 N-tiles of 16x16.
// =====================================================================

// A-fragment (16x32 f16): lanes 0-15 -> M=lane, k-base 0; lanes 16-31 -> M=lane-16, k-base 8.
// Halves: [kb..kb+7] in v0..v3, [kb+16..kb+23] in v4..v7.
static __device__ inline v16h load_a_frag(const _Float16* __restrict__ row, int lane)
{
    int kb = (lane < 16) ? 0 : 8;
    v8h lo = *(const v8h*)(row + kb);
    v8h hi = *(const v8h*)(row + kb + 16);
    v16h r;
#pragma unroll
    for (int j = 0; j < 8; ++j) { r[j] = lo[j]; r[j + 8] = hi[j]; }
    return r;
}

// B-fragment (32x16 f16): lanes 0-15 -> N=lane, K=0..7,16..23; lanes 16-31 -> N=lane-16, K=8..15,24..31.
// LDS holds Wt tile N-major: [n][32 halves], so each lane reads two 8-half chunks.
static __device__ inline v16h load_b_frag(const _Float16* __restrict__ col, int lane)
{
    int kb = (lane < 16) ? 0 : 8;
    v8h lo = *(const v8h*)(col + kb);
    v8h hi = *(const v8h*)(col + kb + 16);
    v16h r;
#pragma unroll
    for (int j = 0; j < 8; ++j) { r[j] = lo[j]; r[j + 8] = hi[j]; }
    return r;
}

// Generic __shared__ pointer -> raw LDS byte offset (aperture mapping:
// LDS offset is the low 32 bits of the flat address, per CDNA5 ISA 10.2).
static __device__ inline unsigned lds_off(const void* p)
{
    return (unsigned)(uintptr_t)p;
}

// Async copy of 32B (two b128) from global to LDS; instruction immediate
// offset applies to BOTH the global and the LDS address, so one address
// pair covers both chunks.  Tracked by ASYNCcnt.
static __device__ inline void async_copy_32B(unsigned ldsaddr, const _Float16* gaddr)
{
    asm volatile(
        "global_load_async_to_lds_b128 %0, %1, off\n\t"
        "global_load_async_to_lds_b128 %0, %1, off offset:16"
        :
        : "v"(ldsaddr), "v"(gaddr)
        : "memory");
}

static __device__ inline void wait_async_all()
{
    asm volatile("s_wait_asynccnt 0x0" ::: "memory");
}

__global__ __launch_bounds__(256) void kan_gemm_kernel(
    const _Float16* __restrict__ A,   // [BATCH][KAUG]
    const _Float16* __restrict__ Wt,  // [OUT][KAUG]
    float* __restrict__ out)          // [BATCH][OUT]
{
    __shared__ __align__(16) _Float16 lA[2][128 * 32];
    __shared__ __align__(16) _Float16 lB[2][128 * 32];

    const int tid  = threadIdx.x;
    const int wave = tid >> 5;
    const int lane = tid & 31;
    const int wr   = wave >> 2;        // 0..1  -> 64 rows each
    const int wc   = wave & 3;         // 0..3  -> 32 cols each

    const int bm = blockIdx.x * 128;
    const int bn = blockIdx.y * 128;

    // staging coordinates: 2 threads per row, 32 halves per row, 16 halves each
    const int ldr = tid >> 1;                 // 0..127
    const int ldc = (tid & 1) * 16;           // half offset 0 / 16

    const _Float16* gA = A  + (size_t)(bm + ldr) * KAUG + ldc;
    const _Float16* gB = Wt + (size_t)(bn + ldr) * KAUG + ldc;

    const unsigned laOff[2] = { lds_off(&lA[0][ldr * 32 + ldc]),
                                lds_off(&lA[1][ldr * 32 + ldc]) };
    const unsigned lbOff[2] = { lds_off(&lB[0][ldr * 32 + ldc]),
                                lds_off(&lB[1][ldr * 32 + ldc]) };

    v8f c[4][2];
#pragma unroll
    for (int mt = 0; mt < 4; ++mt)
#pragma unroll
        for (int nt = 0; nt < 2; ++nt)
            c[mt][nt] = (v8f)(0.0f);

    // ---- preload k-chunk 0 into buffer 0 (async, no VGPR staging) ----
    async_copy_32B(laOff[0], gA);
    async_copy_32B(lbOff[0], gB);
    wait_async_all();
    __syncthreads();

    for (int kt = 0; kt < KT_STEPS; ++kt) {
        const int buf = kt & 1;

        // ---- kick off async copies for the next k-chunk into the other buffer ----
        if (kt + 1 < KT_STEPS) {
            const int nb = buf ^ 1;
            async_copy_32B(laOff[nb], gA + (size_t)(kt + 1) * 32);
            async_copy_32B(lbOff[nb], gB + (size_t)(kt + 1) * 32);
        }

        // ---- compute on current buffer ----
        v16h af[4], bf[2];
#pragma unroll
        for (int mt = 0; mt < 4; ++mt) {
            int row = wr * 64 + mt * 16 + (lane & 15);
            af[mt] = load_a_frag(&lA[buf][row * 32], lane);
        }
#pragma unroll
        for (int nt = 0; nt < 2; ++nt) {
            int col = wc * 32 + nt * 16 + (lane & 15);
            bf[nt] = load_b_frag(&lB[buf][col * 32], lane);
        }
#pragma unroll
        for (int mt = 0; mt < 4; ++mt)
#pragma unroll
            for (int nt = 0; nt < 2; ++nt)
                c[mt][nt] = __builtin_amdgcn_wmma_f32_16x16x32_f16(
                    /*neg_a=*/false, af[mt],
                    /*neg_b=*/false, bf[nt],
                    /*c_mod=*/(short)0, c[mt][nt],
                    /*reuse_a=*/false, /*reuse_b=*/false);

        // own async copies must have landed before anyone reads the other
        // buffer next iteration; barrier makes it workgroup-wide.
        wait_async_all();
        __syncthreads();
    }

    // ---- epilogue: C/D layout -> lanes 0-15: M=r, lanes 16-31: M=8+r; N=lane%16 ----
#pragma unroll
    for (int mt = 0; mt < 4; ++mt) {
#pragma unroll
        for (int nt = 0; nt < 2; ++nt) {
            int m0 = bm + wr * 64 + mt * 16 + ((lane >= 16) ? 8 : 0);
            int n  = bn + wc * 32 + nt * 16 + (lane & 15);
#pragma unroll
            for (int r = 0; r < 8; ++r)
                out[(size_t)(m0 + r) * OUT_DIM + n] = c[mt][nt][r];
        }
    }
}

// =====================================================================
// Host launcher
// =====================================================================
extern "C" void kernel_launch(void* const* d_in, const int* in_sizes, int n_in,
                              void* d_out, int out_size, void* d_ws, size_t ws_size,
                              hipStream_t stream)
{
    const float* x          = (const float*)d_in[0];   // [16384][512]
    const float* grid       = (const float*)d_in[1];   // [512][12]
    const float* coef       = (const float*)d_in[2];   // [512][1024][8]
    const float* scale_base = (const float*)d_in[3];   // [512][1024]
    float* out = (float*)d_out;                        // [16384][1024]

    _Float16* A  = (_Float16*)d_ws;                                        // 16384*4608*2 B
    _Float16* Wt = (_Float16*)((char*)d_ws + (size_t)BATCH * KAUG * 2);    // 1024*4608*2 B

    // 1) activations
    {
        int total = BATCH * IN_DIM;
        kan_act_kernel<<<(total + 255) / 256, 256, 0, stream>>>(x, grid, A);
    }
    // 2) weight pack (tiny: 4.7M elems)
    {
        int total = OUT_DIM * KAUG;
        kan_pack_kernel<<<(total + 255) / 256, 256, 0, stream>>>(coef, scale_base, Wt);
    }
    // 3) WMMA GEMM
    {
        dim3 g(BATCH / 128, OUT_DIM / 128);   // (128, 8)
        kan_gemm_kernel<<<g, 256, 0, stream>>>(A, Wt, out);
    }
}
